// FLD_83236466197026
// MI455X (gfx1250) — matrix operand order repaired and checked by reading, not compiled, not run on hardware
//
#include <hip/hip_runtime.h>
#include <math.h>

typedef unsigned short ushort_t;
typedef __attribute__((ext_vector_type(16))) __bf16 v16bf;
typedef __attribute__((ext_vector_type(8)))  float  v8f;

union frag { v16bf v; ushort_t u[16]; };

#define WMMA_BF16(A, B, C) \
  __builtin_amdgcn_wmma_f32_16x16x32_bf16(false, (A), false, (B), (short)0, (C), false, false)

#define B_DIM 64
#define L_DIM 2048
#define T_DIM 1024
#define D_DIM 128
#define E_DIM 512
#define H_DIM 8
#define P_DIM 3
#define LATENT 256
#define HIDDEN 512

__device__ __forceinline__ v8f zero8() { v8f z = {}; return z; }

// f32 -> bf16 round-to-nearest-even
__device__ __forceinline__ ushort_t f2bf(float f) {
  unsigned int u = __float_as_uint(f);
  u += 0x7FFFu + ((u >> 16) & 1u);
  return (ushort_t)(u >> 16);
}

// async 16-byte copy global -> LDS (ASYNCcnt-tracked)
__device__ __forceinline__ void async_copy16(unsigned lds_off, const void* gptr) {
  asm volatile("global_load_async_to_lds_b128 %0, %1, off"
               :: "v"(lds_off), "v"(gptr) : "memory");
}
__device__ __forceinline__ void wait_async0() {
  asm volatile("s_wait_asynccnt 0" ::: "memory");
}

// A fragment (16x32 bf16): lane holds row m=lane&15; kb8 = (lane>>4)*8.
// VGPR i holds K = 2*kb8 + 2*i (pair).
__device__ __forceinline__ void load_a(frag& f, const ushort_t* row, int kb8) {
#pragma unroll
  for (int i = 0; i < 8; ++i) {
    int k0 = 2 * kb8 + 2 * i;
    f.u[2 * i]     = row[k0];
    f.u[2 * i + 1] = row[k0 + 1];
  }
}

// B fragment (32x16 bf16): lane holds col n=lane&15; kb16=(lane>>4)*16.
// caller passes (row_n_base + kb16 + kstep*32); pairs contiguous.
__device__ __forceinline__ void load_b(frag& f, const ushort_t* row) {
#pragma unroll
  for (int i = 0; i < 8; ++i) {
    f.u[2 * i]     = row[2 * i];
    f.u[2 * i + 1] = row[2 * i + 1];
  }
}

// ---------------------------------------------------------------- prep
// src[k*N + n] (f32, K x N) -> dst[n*K + k] (bf16, N x K)
__global__ __launch_bounds__(256) void transpose_bf16_kernel(
    const float* __restrict__ src, ushort_t* __restrict__ dst, int K, int N) {
  int idx = blockIdx.x * 256 + threadIdx.x;
  if (idx >= K * N) return;
  int k = idx / N, n = idx % N;
  dst[n * K + k] = f2bf(src[k * N + n]);
}

// q = query @ W_q + b_q  (3x512), stored as qT[h][p_pad16][64] bf16, p>=3 -> 0
__global__ __launch_bounds__(256) void q_kernel(
    const float* __restrict__ query, const float* __restrict__ Wq,
    const float* __restrict__ bq, ushort_t* __restrict__ qT) {
  int idx = blockIdx.x * 256 + threadIdx.x;      // 8*16*64 = 8192
  if (idx >= H_DIM * 16 * 64) return;
  int h = idx / (16 * 64);
  int p = (idx / 64) % 16;
  int e = idx % 64;
  int col = h * 64 + e;
  float acc = 0.f;
  if (p < P_DIM) {
    acc = bq[col];
    for (int i = 0; i < E_DIM; ++i) acc += query[p * E_DIM + i] * Wq[i * E_DIM + col];
  }
  qT[idx] = f2bf(acc);
}

// ------------------------------------------------- fused k-GEMM + scores
// per wave: 16 rows of l. key generated on the fly; Wk_t B-panels staged
// block-cooperatively in LDS via double-buffered async copies; k tile staged
// in LDS; scores = (k_h . q_h^T)/8 written to ws.
__global__ __launch_bounds__(256) void k_scores_kernel(
    const float* __restrict__ timesteps, const float* __restrict__ w_te,
    const float* __restrict__ b_te, const ushort_t* __restrict__ Wk_t,
    const float* __restrict__ b_k, const ushort_t* __restrict__ qT,
    float* __restrict__ scores) {
  __shared__ float te_w[E_DIM], te_b[E_DIM];
  __shared__ ushort_t kstage[8][16][32];
  __shared__ ushort_t panels[2][16][E_DIM];     // double-buffered B panels (16 cols x 512 K)
  int tid = threadIdx.x;
  for (int i = tid; i < E_DIM; i += 256) { te_w[i] = w_te[i]; te_b[i] = b_te[i]; }

  int b    = blockIdx.x >> 4;      // 16 l-tiles of 128 per batch
  int lt   = blockIdx.x & 15;
  int wave = tid >> 5, lane = tid & 31;
  int l0   = lt * 128 + wave * 16;
  int mrow = lane & 15;
  int kb8  = (lane >> 4) * 8;
  int kb16 = (lane >> 4) * 16;
  int n4   = lane & 15;
  int mlo  = 8 * (lane >> 4);
  float t  = timesteps[b * L_DIM + l0 + mrow];

  // panel DMA: 16 rows x 1KB = 1024 x 16B transfers, 4 per thread
  auto issue_panel = [&](int nt, int bufi) {
    int ncol = nt * 16;
#pragma unroll
    for (int j = 0; j < 4; ++j) {
      int idx = tid + j * 256;     // 0..1023
      int rw  = idx >> 6;          // panel row (column of Wk_t)
      int o16 = idx & 63;          // 16B chunk within 1KB row
      unsigned lds_off = (unsigned)(size_t)&panels[bufi][rw][o16 * 8];
      const ushort_t* g = Wk_t + (size_t)(ncol + rw) * E_DIM + o16 * 8;
      async_copy16(lds_off, g);
    }
  };

  issue_panel(0, 0);

  // hoist key A-fragments for all 16 K-steps (E=512) while DMA flies
  __syncthreads();    // te_w/te_b ready
  frag af[16];
#pragma unroll
  for (int ks = 0; ks < 16; ++ks) {
#pragma unroll
    for (int i = 0; i < 8; ++i) {
      int kloc = 2 * kb8 + 2 * i;
      int kg   = ks * 32 + kloc;
      float a0 = fmaf(t, te_w[kg], te_b[kg]);
      if ((i & 3) == 0) a0 = sinf(a0);     // exactly the (col % 8 == 0) columns
      float a1 = fmaf(t, te_w[kg + 1], te_b[kg + 1]);
      af[ks].u[2 * i]     = f2bf(a0);
      af[ks].u[2 * i + 1] = f2bf(a1);
    }
  }
  wait_async0();
  __syncthreads();    // panel 0 visible to all waves

  v8f sacc = zero8();
  for (int nt = 0; nt < 32; ++nt) {          // nt = h*4 + estep*2 + s
    int h = nt >> 2, estep = (nt >> 1) & 1, s = nt & 1;
    if ((nt & 3) == 0) sacc = zero8();
    if (nt + 1 < 32) issue_panel(nt + 1, (nt + 1) & 1);   // overlap DMA with WMMA

    v8f kacc = zero8();
    const ushort_t* Bb = &panels[nt & 1][n4][kb16];
#pragma unroll
    for (int ks = 0; ks < 16; ++ks) {
      frag bf_;
      load_b(bf_, Bb + ks * 32);
      kacc = WMMA_BF16(af[ks].v, bf_.v, kacc);
    }
    float bk = b_k[nt * 16 + n4];
#pragma unroll
    for (int r = 0; r < 8; ++r)
      kstage[wave][r + mlo][s * 16 + n4] = f2bf(kacc[r] + bk);

    if (s == 1) {
      // scores WMMA: A = staged k (16x32), B = qT for head h
      frag a2, b2;
      load_a(a2, &kstage[wave][mrow][0], kb8);
      load_b(b2, qT + (h * 16 + n4) * 64 + estep * 32 + kb16);
      sacc = WMMA_BF16(a2.v, b2.v, sacc);
      if (estep == 1 && n4 < P_DIM) {
#pragma unroll
        for (int r = 0; r < 8; ++r)
          scores[(b * 24 + h * 3 + n4) * L_DIM + l0 + r + mlo] = sacc[r] * 0.125f;
      }
    }
    wait_async0();      // my copies for nt+1 landed
    __syncthreads();    // everyone done reading panel nt & copies complete
  }
}

// ------------------------------------- row max + exp -> bf16 weights
__global__ __launch_bounds__(256) void softmax_w_kernel(
    const float* __restrict__ scores, ushort_t* __restrict__ wbf) {
  __shared__ float red[256];
  int row = blockIdx.x;                 // b*24 + hp
  int b = row / 24, hp = row % 24;
  int tid = threadIdx.x;
  const float* s = scores + (size_t)row * L_DIM;
  float mx = -INFINITY;
  for (int l = tid; l < L_DIM; l += 256) mx = fmaxf(mx, s[l]);
  red[tid] = mx;
  __syncthreads();
  for (int st = 128; st; st >>= 1) {
    if (tid < st) red[tid] = fmaxf(red[tid], red[tid + st]);
    __syncthreads();
  }
  mx = red[0];
  ushort_t* w = wbf + ((size_t)b * 32 + hp) * L_DIM;   // padded to 32 rows/batch
  for (int l = tid; l < L_DIM; l += 256) w[l] = f2bf(expf(s[l] - mx));
}

// ------------------------------------- attention weighted sums + x
// per block: one batch.  S = w(24x2048) @ [M*X, M](2048x256), x = [S1/S2, 1]
__global__ __launch_bounds__(256) void attn_kernel(
    const float* __restrict__ X, const float* __restrict__ M,
    const ushort_t* __restrict__ wbf, ushort_t* __restrict__ xbf) {
  __shared__ ushort_t Vt[256][32];
  __shared__ float Sres[32][256];
  int b = blockIdx.x;
  int tid = threadIdx.x, wave = tid >> 5, lane = tid & 31;
  int mtile = wave >> 2, ng = wave & 3;
  int n4 = lane & 15, kb16 = (lane >> 4) * 16, kb8 = (lane >> 4) * 8;
  int mlo = 8 * (lane >> 4);
  v8f acc[4] = {{}, {}, {}, {}};
  int hp = mtile * 16 + n4;
  const ushort_t* Ab = wbf + ((size_t)b * 32 + hp) * L_DIM;

  for (int ks = 0; ks < 64; ++ks) {
    __syncthreads();
    for (int idx = tid; idx < 32 * 256; idx += 256) {
      int lloc = idx >> 8, c = idx & 255;
      int l = ks * 32 + lloc;
      float mv = M[((size_t)b * L_DIM + l) * D_DIM + (c & 127)];
      float v  = (c < 128) ? mv * X[((size_t)b * L_DIM + l) * D_DIM + c] : mv;
      Vt[c][lloc] = f2bf(v);
    }
    // prefetch next chunk of X/M (the dominant HBM stream)
    if (ks + 1 < 64) {
      int lp = (ks + 1) * 32 + (tid >> 3);
      const size_t base = ((size_t)b * L_DIM + lp) * D_DIM + (size_t)(tid & 7) * 16;
      __builtin_prefetch(&X[base], 0, 0);
      __builtin_prefetch(&M[base], 0, 0);
    }
    __syncthreads();
    frag a;
    load_a(a, Ab + ks * 32, kb8);
#pragma unroll
    for (int j = 0; j < 4; ++j) {
      int nt = ng * 4 + j;
      frag bfr;
      load_b(bfr, &Vt[nt * 16 + n4][kb16]);
      acc[j] = WMMA_BF16(a.v, bfr.v, acc[j]);
    }
  }
#pragma unroll
  for (int j = 0; j < 4; ++j) {
    int nt = ng * 4 + j;
#pragma unroll
    for (int r = 0; r < 8; ++r) Sres[mtile * 16 + r + mlo][nt * 16 + n4] = acc[j][r];
  }
  __syncthreads();
  for (int idx = tid; idx < 24 * 256; idx += 256) {
    int hh = idx >> 8, c = idx & 255;
    int h = hh / 3, p = hh % 3;
    float v = (c < 128) ? (Sres[hh][c] / Sres[hh][128 + c]) : 1.0f;
    xbf[((size_t)(b * 3 + p)) * 2048 + h * 256 + c] = f2bf(v);
  }
}

// ------------------------------------- coeffs = x @ W_o + b_o  (192x2048 @ 2048x256)
__global__ __launch_bounds__(256) void coeffs_kernel(
    const ushort_t* __restrict__ xbf, const ushort_t* __restrict__ Wo_t,
    const float* __restrict__ b_o, float* __restrict__ coeffs) {
  int tid = threadIdx.x, wave = tid >> 5, lane = tid & 31;
  int tile = blockIdx.x * 8 + wave;           // 0..191
  int mtile = tile >> 4, ntile = tile & 15;
  int n4 = lane & 15, kb16 = (lane >> 4) * 16, kb8 = (lane >> 4) * 8;
  int mlo = 8 * (lane >> 4);
  const ushort_t* Ab = xbf + (size_t)(mtile * 16 + n4) * 2048;
  const ushort_t* Bb = Wo_t + (size_t)(ntile * 16 + n4) * 2048 + kb16;
  v8f acc = zero8();
  for (int ks = 0; ks < 64; ++ks) {
    frag a, bfr;
    load_a(a, Ab + ks * 32, kb8);
    load_b(bfr, Bb + ks * 32);
    acc = WMMA_BF16(a.v, bfr.v, acc);
  }
  float bo = b_o[ntile * 16 + n4];
#pragma unroll
  for (int r = 0; r < 8; ++r)
    coeffs[(size_t)(mtile * 16 + r + mlo) * LATENT + ntile * 16 + n4] = acc[r] + bo;
}

// ------------------------------------- fused z -> relu(zW1) -> relu(h1W2) -> h2W3
// one wave per 16-row tile; intermediates stay in LDS.
__global__ __launch_bounds__(32) void mlp_kernel(
    const float* __restrict__ coeffs, const float* __restrict__ yts,
    const ushort_t* __restrict__ W1t, const float* __restrict__ b1,
    const ushort_t* __restrict__ W2t, const float* __restrict__ b2,
    const ushort_t* __restrict__ W3t, const float* __restrict__ b3,
    float* __restrict__ out) {
  __shared__ ushort_t zb[16][256];
  __shared__ ushort_t h1[16][512];
  __shared__ ushort_t h2[16][512];
  int lane = threadIdx.x;
  int row0 = blockIdx.x * 16;
  int b = row0 >> 10;                       // T = 1024, 16 | 1024
  for (int idx = lane; idx < 16 * 256; idx += 32) {
    int m = idx >> 8, c = idx & 255;
    float tt = yts[row0 + m];
    const float* cb = coeffs + (size_t)(b * 3) * 256;
    float z = cb[c] + tt * cb[256 + c] + tt * tt * cb[512 + c];
    zb[m][c] = f2bf(z);
  }
  int n4 = lane & 15, kb16 = (lane >> 4) * 16, kb8 = (lane >> 4) * 8;
  int mlo = 8 * (lane >> 4), mrow = lane & 15;

  // layer 1: 16x256 @ 256x512
  frag af[8];
#pragma unroll
  for (int ks = 0; ks < 8; ++ks) load_a(af[ks], &zb[mrow][ks * 32], kb8);
  for (int nt = 0; nt < 32; ++nt) {
    v8f acc = zero8();
    const ushort_t* Bb = W1t + (size_t)(nt * 16 + n4) * 256 + kb16;
#pragma unroll
    for (int ks = 0; ks < 8; ++ks) {
      frag bfr; load_b(bfr, Bb + ks * 32);
      acc = WMMA_BF16(af[ks].v, bfr.v, acc);
    }
    float bb = b1[nt * 16 + n4];
#pragma unroll
    for (int r = 0; r < 8; ++r)
      h1[r + mlo][nt * 16 + n4] = f2bf(fmaxf(acc[r] + bb, 0.f));
  }

  // layer 2: 16x512 @ 512x512
  frag ag[16];
#pragma unroll
  for (int ks = 0; ks < 16; ++ks) load_a(ag[ks], &h1[mrow][ks * 32], kb8);
  for (int nt = 0; nt < 32; ++nt) {
    v8f acc = zero8();
    const ushort_t* Bb = W2t + (size_t)(nt * 16 + n4) * 512 + kb16;
#pragma unroll
    for (int ks = 0; ks < 16; ++ks) {
      frag bfr; load_b(bfr, Bb + ks * 32);
      acc = WMMA_BF16(ag[ks].v, bfr.v, acc);
    }
    float bb = b2[nt * 16 + n4];
#pragma unroll
    for (int r = 0; r < 8; ++r)
      h2[r + mlo][nt * 16 + n4] = f2bf(fmaxf(acc[r] + bb, 0.f));
  }

  // layer 3: 16x512 @ 512x128 -> global out
  frag ah[16];
#pragma unroll
  for (int ks = 0; ks < 16; ++ks) load_a(ah[ks], &h2[mrow][ks * 32], kb8);
  for (int nt = 0; nt < 8; ++nt) {
    v8f acc = zero8();
    const ushort_t* Bb = W3t + (size_t)(nt * 16 + n4) * 512 + kb16;
#pragma unroll
    for (int ks = 0; ks < 16; ++ks) {
      frag bfr; load_b(bfr, Bb + ks * 32);
      acc = WMMA_BF16(ah[ks].v, bfr.v, acc);
    }
    float bb = b3[nt * 16 + n4];
#pragma unroll
    for (int r = 0; r < 8; ++r)
      out[(size_t)(row0 + r + mlo) * D_DIM + nt * 16 + n4] = acc[r] + bb;
  }
}

// ---------------------------------------------------------------- launch
extern "C" void kernel_launch(void* const* d_in, const int* in_sizes, int n_in,
                              void* d_out, int out_size, void* d_ws, size_t ws_size,
                              hipStream_t stream) {
  const float* timesteps = (const float*)d_in[0];
  const float* X     = (const float*)d_in[1];
  const float* M     = (const float*)d_in[2];
  const float* yts   = (const float*)d_in[3];
  const float* w_te  = (const float*)d_in[4];
  const float* b_te  = (const float*)d_in[5];
  const float* query = (const float*)d_in[6];
  const float* W_q   = (const float*)d_in[7];
  const float* b_q   = (const float*)d_in[8];
  const float* W_k   = (const float*)d_in[9];
  const float* b_k   = (const float*)d_in[10];
  const float* W_o   = (const float*)d_in[11];
  const float* b_o   = (const float*)d_in[12];
  const float* W1    = (const float*)d_in[13];
  const float* b1    = (const float*)d_in[14];
  const float* W2    = (const float*)d_in[15];
  const float* b2    = (const float*)d_in[16];
  const float* W3    = (const float*)d_in[17];
  const float* b3    = (const float*)d_in[18];
  float* out = (float*)d_out;

  char* ws = (char*)d_ws;
  const size_t OFF_WKT    = 0;                 // 512*512*2    = 524288
  const size_t OFF_WOT    = 524288;            // 256*2048*2   = 1048576
  const size_t OFF_W1T    = 1572864;           // 512*256*2    = 262144
  const size_t OFF_W2T    = 1835008;           // 512*512*2    = 524288
  const size_t OFF_W3T    = 2359296;           // 128*512*2    = 131072
  const size_t OFF_QT     = 2490368;           // 8*16*64*2    = 16384
  const size_t OFF_SCORES = 2506752;           // 64*24*2048*4 = 12582912
  const size_t OFF_WBF    = 15089664;          // 64*32*2048*2 = 8388608
  const size_t OFF_XBF    = 23478272;          // 192*2048*2   = 786432
  const size_t OFF_COEF   = 24264704;          // 192*256*4    = 196608

  ushort_t* Wk_t = (ushort_t*)(ws + OFF_WKT);
  ushort_t* Wo_t = (ushort_t*)(ws + OFF_WOT);
  ushort_t* W1_t = (ushort_t*)(ws + OFF_W1T);
  ushort_t* W2_t = (ushort_t*)(ws + OFF_W2T);
  ushort_t* W3_t = (ushort_t*)(ws + OFF_W3T);
  ushort_t* qT   = (ushort_t*)(ws + OFF_QT);
  float*    scores = (float*)(ws + OFF_SCORES);
  ushort_t* wbf  = (ushort_t*)(ws + OFF_WBF);
  ushort_t* xbf  = (ushort_t*)(ws + OFF_XBF);
  float*    coeffs = (float*)(ws + OFF_COEF);

  // weight transposes to bf16 [N][K]
  transpose_bf16_kernel<<<(512 * 512 + 255) / 256, 256, 0, stream>>>(W_k, Wk_t, 512, 512);
  transpose_bf16_kernel<<<(2048 * 256 + 255) / 256, 256, 0, stream>>>(W_o, Wo_t, 2048, 256);
  transpose_bf16_kernel<<<(256 * 512 + 255) / 256, 256, 0, stream>>>(W1, W1_t, 256, 512);
  transpose_bf16_kernel<<<(512 * 512 + 255) / 256, 256, 0, stream>>>(W2, W2_t, 512, 512);
  transpose_bf16_kernel<<<(512 * 128 + 255) / 256, 256, 0, stream>>>(W3, W3_t, 512, 128);

  q_kernel<<<(8 * 16 * 64 + 255) / 256, 256, 0, stream>>>(query, W_q, b_q, qT);

  k_scores_kernel<<<B_DIM * 16, 256, 0, stream>>>(timesteps, w_te, b_te, Wk_t, b_k, qT, scores);

  softmax_w_kernel<<<B_DIM * 24, 256, 0, stream>>>(scores, wbf);

  attn_kernel<<<B_DIM, 256, 0, stream>>>(X, M, wbf, xbf);

  coeffs_kernel<<<24, 256, 0, stream>>>(xbf, Wo_t, b_o, coeffs);

  mlp_kernel<<<(B_DIM * T_DIM) / 16, 32, 0, stream>>>(coeffs, yts, W1_t, b1, W2_t, b2, W3_t, b3, out);
}
